// GEncoder2_32753420600135
// MI455X (gfx1250) — compile-verified, matrix operand order checked by reading, not run on hardware
//
#include <hip/hip_runtime.h>

typedef __attribute__((ext_vector_type(16))) __bf16        v16bf;
typedef __attribute__((ext_vector_type(8)))  float         v8f;
typedef __attribute__((ext_vector_type(8)))  unsigned int  v8u;

// ---------------- constants (reference is fixed-shape) ----------------
#define C_N   20000
#define C_E   160000
#define C_B   100
#define C_M   200
#define C_H   512
#define C_IN  177
#define C_INP 192      // 177 padded to mult of 32
#define C_ED  16
#define C_P   768
#define C_L   3
#define C_BPAD 128     // 100 head rows padded to mult of 32

// ---------------- helpers ----------------
__device__ __forceinline__ unsigned short f2bf(float f) {
  unsigned u = __float_as_uint(f);
  u += 0x7fffu + ((u >> 16) & 1u);     // round to nearest even
  return (unsigned short)(u >> 16);
}
__device__ __forceinline__ unsigned fkey(float f) {          // order-preserving key
  unsigned u = __float_as_uint(f);
  return (u & 0x80000000u) ? ~u : (u | 0x80000000u);
}
__device__ __forceinline__ float fdec(unsigned k) {
  unsigned u = (k & 0x80000000u) ? (k & 0x7fffffffu) : ~k;
  return __uint_as_float(u);
}

// ---------------- fp32 -> bf16, row-padded / K-padded ----------------
__global__ __launch_bounds__(256) void cvt_bf16_pad(const float* __restrict__ src,
                                                    unsigned short* __restrict__ dst,
                                                    int rows, int K, int rowsPad, int Kpad) {
  long long idx = (long long)blockIdx.x * 256 + threadIdx.x;
  long long total = (long long)rowsPad * Kpad;
  if (idx >= total) return;
  int r = (int)(idx / Kpad), k = (int)(idx % Kpad);
  float f = (r < rows && k < K) ? src[(long long)r * K + k] : 0.f;
  dst[idx] = f2bf(f);
}

// weight [K,N] fp32 -> transposed bf16 [N,Kpad]
__global__ __launch_bounds__(256) void cvtT_bf16_pad(const float* __restrict__ src,
                                                     unsigned short* __restrict__ dst,
                                                     int K, int N, int Kpad) {
  long long idx = (long long)blockIdx.x * 256 + threadIdx.x;
  long long total = (long long)N * Kpad;
  if (idx >= total) return;
  int n = (int)(idx / Kpad), k = (int)(idx % Kpad);
  float f = (k < K) ? src[(long long)k * N + n] : 0.f;
  dst[idx] = f2bf(f);
}

// ---- WMMA bf16 GEMM: C[M,N] = A[Mpad,Kpad] @ Wt[N,Kpad]^T + bias --------------
// One wave -> 32x32 output block (2x2 register-blocked 16x16 WMMA tiles).
// Each 32-K step: 2 A-fragments + 2 B-fragments feed 4 v_wmma, f32 accumulate.
__global__ __launch_bounds__(128) void wmma_gemm_bf16(const unsigned short* __restrict__ A,
                                                      const unsigned short* __restrict__ Wt,
                                                      const float* __restrict__ bias,
                                                      float* __restrict__ C,
                                                      int M, int Mpad, int N, int Kpad,
                                                      int relu) {
  int lane   = threadIdx.x;                 // 0..31
  int blk_m  = blockIdx.y * 4 + threadIdx.y;
  int blk_n  = blockIdx.x;
  if (blk_m * 32 >= Mpad) return;           // wave-uniform exit, EXEC stays all-ones
  int half = lane >> 4;                     // ISA 16-bit A layout: lanes16-31 -> K+8
  int l15  = lane & 15;

  const unsigned* Ar0 = (const unsigned*)(A  + (long long)(blk_m * 32 + l15) * Kpad);
  const unsigned* Ar1 = (const unsigned*)(A  + (long long)(blk_m * 32 + 16 + l15) * Kpad);
  const unsigned* Br0 = (const unsigned*)(Wt + (long long)(blk_n * 32 + l15) * Kpad);
  const unsigned* Br1 = (const unsigned*)(Wt + (long long)(blk_n * 32 + 16 + l15) * Kpad);

  v8f acc00 = {}, acc01 = {}, acc10 = {}, acc11 = {};
  for (int k0 = 0; k0 < Kpad; k0 += 32) {
    int i0 = (k0 + half * 8) >> 1;          // dword index of first K pair
    int i1 = i0 + 8;                        // (k0+16+half*8)/2
    v8u a0u, a1u, b0u, b1u;
#pragma unroll
    for (int v = 0; v < 4; ++v) {
      a0u[v] = Ar0[i0 + v]; a0u[4 + v] = Ar0[i1 + v];
      a1u[v] = Ar1[i0 + v]; a1u[4 + v] = Ar1[i1 + v];
      b0u[v] = Br0[i0 + v]; b0u[4 + v] = Br0[i1 + v];
      b1u[v] = Br1[i0 + v]; b1u[4 + v] = Br1[i1 + v];
    }
    v16bf a0 = __builtin_bit_cast(v16bf, a0u);
    v16bf a1 = __builtin_bit_cast(v16bf, a1u);
    v16bf b0 = __builtin_bit_cast(v16bf, b0u);
    v16bf b1 = __builtin_bit_cast(v16bf, b1u);
    acc00 = __builtin_amdgcn_wmma_f32_16x16x32_bf16(false, a0, false, b0, (short)0, acc00, false, false);
    acc01 = __builtin_amdgcn_wmma_f32_16x16x32_bf16(false, a0, false, b1, (short)0, acc01, false, false);
    acc10 = __builtin_amdgcn_wmma_f32_16x16x32_bf16(false, a1, false, b0, (short)0, acc10, false, false);
    acc11 = __builtin_amdgcn_wmma_f32_16x16x32_bf16(false, a1, false, b1, (short)0, acc11, false, false);
  }

  int col0 = blk_n * 32 + l15;
  int col1 = col0 + 16;
  float bv0 = bias ? bias[col0] : 0.f;
  float bv1 = bias ? bias[col1] : 0.f;
#pragma unroll
  for (int r = 0; r < 8; ++r) {
    int row0 = blk_m * 32 + half * 8 + r;   // C/D layout: lanes16-31 -> M+8
    int row1 = row0 + 16;
    if (row0 < M) {
      float v00 = acc00[r] + bv0, v01 = acc01[r] + bv1;
      if (relu) { v00 = fmaxf(v00, 0.f); v01 = fmaxf(v01, 0.f); }
      C[(long long)row0 * N + col0] = v00;
      C[(long long)row0 * N + col1] = v01;
    }
    if (row1 < M) {
      float v10 = acc10[r] + bv0, v11 = acc11[r] + bv1;
      if (relu) { v10 = fmaxf(v10, 0.f); v11 = fmaxf(v11, 0.f); }
      C[(long long)row1 * N + col0] = v10;
      C[(long long)row1 * N + col1] = v11;
    }
  }
}

// ---------------- edge pass 1: s = leaky_relu(hs[src]+hd[dst]+he) . a_vec ------
__global__ __launch_bounds__(128) void edge_score(const int* __restrict__ ei,
                                                  const float* __restrict__ eattr,
                                                  const float* __restrict__ hs,
                                                  const float* __restrict__ hd,
                                                  const float* __restrict__ we,
                                                  const float* __restrict__ be,
                                                  const float* __restrict__ avec,
                                                  float* __restrict__ sE,
                                                  unsigned* __restrict__ smax) {
  int e = blockIdx.x * 4 + threadIdx.y;
  if (e >= C_E) return;
  int lane = threadIdx.x;
  int s = ei[e], d = ei[C_E + e];
  float ea[C_ED];
#pragma unroll
  for (int t = 0; t < C_ED; ++t) ea[t] = eattr[(long long)e * C_ED + t];
  float part = 0.f;
  for (int j = 0; j < C_H / 32; ++j) {
    int h = j * 32 + lane;
    float he = be[h];
#pragma unroll
    for (int t = 0; t < C_ED; ++t) he += ea[t] * we[t * C_H + h];
    float m = hs[(long long)s * C_H + h] + hd[(long long)d * C_H + h] + he;
    float lr = m > 0.f ? m : 0.2f * m;
    part += lr * avec[h];
  }
  for (int off = 16; off >= 1; off >>= 1) part += __shfl_down(part, off, 32);
  if (lane == 0) {
    sE[e] = part;
    atomicMax(smax + d, fkey(part));
  }
}

// ---------------- edge pass 2: ex = exp(s - smax[dst]); den[dst] += ex ---------
__global__ __launch_bounds__(256) void edge_exp(const int* __restrict__ ei,
                                                float* __restrict__ sE,
                                                const unsigned* __restrict__ smax,
                                                float* __restrict__ den) {
  int e = blockIdx.x * 256 + threadIdx.x;
  if (e >= C_E) return;
  int d = ei[C_E + e];
  float ex = __expf(sE[e] - fdec(smax[d]));
  sE[e] = ex;
  atomicAdd(den + d, ex);
}

// ---------------- edge pass 3: xnew[dst] += alpha * (hs[src] + he) -------------
__global__ __launch_bounds__(128) void edge_scatter(const int* __restrict__ ei,
                                                    const float* __restrict__ eattr,
                                                    const float* __restrict__ hs,
                                                    const float* __restrict__ we,
                                                    const float* __restrict__ be,
                                                    const float* __restrict__ sE,
                                                    const float* __restrict__ den,
                                                    float* __restrict__ xnew) {
  int e = blockIdx.x * 4 + threadIdx.y;
  if (e >= C_E) return;
  int lane = threadIdx.x;
  int s = ei[e], d = ei[C_E + e];
  float alpha = sE[e] / (den[d] + 1e-9f);
  float ea[C_ED];
#pragma unroll
  for (int t = 0; t < C_ED; ++t) ea[t] = eattr[(long long)e * C_ED + t];
  for (int j = 0; j < C_H / 32; ++j) {
    int h = j * 32 + lane;
    float he = be[h];
#pragma unroll
    for (int t = 0; t < C_ED; ++t) he += ea[t] * we[t * C_H + h];
    atomicAdd(&xnew[(long long)d * C_H + h],
              alpha * (hs[(long long)s * C_H + h] + he));
  }
}

// ---------------- x = rms(x + delta) * w  (optional relu, in place) ------------
__global__ __launch_bounds__(256) void rms_kernel(float* __restrict__ x,
                                                  const float* __restrict__ delta,
                                                  const float* __restrict__ w,
                                                  int cols, int relu) {
  int row = blockIdx.x, tid = threadIdx.x;
  long long base = (long long)row * cols;
  float ss = 0.f;
  for (int j = tid; j < cols; j += 256) {
    float y = x[base + j] + (delta ? delta[base + j] : 0.f);
    ss += y * y;
  }
  __shared__ float red[256];
  red[tid] = ss;
  __syncthreads();
  for (int s = 128; s > 0; s >>= 1) {
    if (tid < s) red[tid] += red[tid + s];
    __syncthreads();
  }
  float scale = rsqrtf(red[0] / (float)cols + 1.1920929e-07f);
  for (int j = tid; j < cols; j += 256) {
    float y = x[base + j] + (delta ? delta[base + j] : 0.f);
    y = y * scale * w[j];
    if (relu) y = fmaxf(y, 0.f);
    x[base + j] = y;
  }
}

// ---------------- att[i] = z[i] . av_w + av_b ----------------------------------
__global__ __launch_bounds__(128) void att_kernel(const float* __restrict__ z,
                                                  const float* __restrict__ avw,
                                                  const float* __restrict__ avb,
                                                  float* __restrict__ att) {
  int row = blockIdx.x * 4 + threadIdx.y;
  if (row >= C_N) return;
  int lane = threadIdx.x;
  float part = 0.f;
  for (int j = 0; j < C_H / 32; ++j) {
    int h = j * 32 + lane;
    part += z[(long long)row * C_H + h] * avw[h];
  }
  for (int off = 16; off >= 1; off >>= 1) part += __shfl_down(part, off, 32);
  if (lane == 0) att[row] = part + avb[0];
}

// ---------------- per-graph softmax pooling: hg[b] = sum_m alpha_m z[b,m] ------
__global__ __launch_bounds__(256) void pool_kernel(const float* __restrict__ z,
                                                   const float* __restrict__ att,
                                                   float* __restrict__ hg) {
  int b = blockIdx.x, tid = threadIdx.x;
  __shared__ float red[256];
  __shared__ float salpha[C_M];
  float mx = -3.4e38f;
  for (int m = tid; m < C_M; m += 256) mx = fmaxf(mx, att[b * C_M + m]);
  red[tid] = mx;
  __syncthreads();
  for (int s = 128; s > 0; s >>= 1) {
    if (tid < s) red[tid] = fmaxf(red[tid], red[tid + s]);
    __syncthreads();
  }
  mx = red[0];
  __syncthreads();
  float sum = 0.f;
  for (int m = tid; m < C_M; m += 256) {
    float e = __expf(att[b * C_M + m] - mx);
    salpha[m] = e;
    sum += e;
  }
  red[tid] = sum;
  __syncthreads();
  for (int s = 128; s > 0; s >>= 1) {
    if (tid < s) red[tid] += red[tid + s];
    __syncthreads();
  }
  float inv = 1.f / red[0];
  __syncthreads();
  for (int h = tid; h < C_H; h += 256) {
    float acc = 0.f;
    for (int m = 0; m < C_M; ++m)
      acc += z[((long long)b * C_M + m) * C_H + h] * salpha[m];
    hg[(long long)b * C_H + h] = acc * inv;
  }
}

// ================================ host ========================================
extern "C" void kernel_launch(void* const* d_in, const int* in_sizes, int n_in,
                              void* d_out, int out_size, void* d_ws, size_t ws_size,
                              hipStream_t stream) {
  const float* x_in  = (const float*)d_in[0];
  const int*   ei    = (const int*)d_in[1];
  const float* eattr = (const float*)d_in[2];
  const float* in_w  = (const float*)d_in[4];
  const float* in_b  = (const float*)d_in[5];
  const float* wsrc  = (const float*)d_in[6];
  const float* bsrc  = (const float*)d_in[7];
  const float* wdst  = (const float*)d_in[8];
  const float* bdst  = (const float*)d_in[9];
  const float* we    = (const float*)d_in[10];
  const float* be    = (const float*)d_in[11];
  const float* avec  = (const float*)d_in[12];
  const float* normw = (const float*)d_in[13];
  const float* fw1   = (const float*)d_in[14];
  const float* fb1   = (const float*)d_in[15];
  const float* fw2   = (const float*)d_in[16];
  const float* fb2   = (const float*)d_in[17];
  const float* gapw  = (const float*)d_in[18];
  const float* gapb  = (const float*)d_in[19];
  const float* avw   = (const float*)d_in[20];
  const float* avb   = (const float*)d_in[21];
  const float* pw1   = (const float*)d_in[22];
  const float* pb1   = (const float*)d_in[23];
  const float* pnw   = (const float*)d_in[24];
  const float* pw2   = (const float*)d_in[25];
  const float* pb2   = (const float*)d_in[26];

  char* ws = (char*)d_ws;
  size_t off = 0;
  auto alloc = [&](size_t bytes) -> char* {
    char* p = ws + off;
    off += (bytes + 255) & ~(size_t)255;
    return p;
  };
  float*          xcur = (float*)alloc((size_t)C_N * C_H * 4);
  float*          hs   = (float*)alloc((size_t)C_N * C_H * 4);
  float*          hd   = (float*)alloc((size_t)C_N * C_H * 4);
  float*          xnew = (float*)alloc((size_t)C_N * C_H * 4);
  unsigned short* xbf  = (unsigned short*)alloc((size_t)C_N * C_H * 2);
  unsigned short* wbf  = (unsigned short*)alloc((size_t)C_P * C_P * 2);
  float*          sE   = (float*)alloc((size_t)C_E * 4);
  unsigned*       smax = (unsigned*)alloc((size_t)C_N * 4);
  float*          den  = (float*)alloc((size_t)C_N * 4);
  float*          att  = (float*)alloc((size_t)C_N * 4);
  float*          hg   = (float*)alloc((size_t)C_BPAD * C_H * 4);
  float*          t1   = (float*)alloc((size_t)C_BPAD * C_P * 4);

  auto cvt = [&](const float* src, unsigned short* dst, int rows, int K, int rowsPad, int Kpad) {
    long long total = (long long)rowsPad * Kpad;
    cvt_bf16_pad<<<(unsigned)((total + 255) / 256), 256, 0, stream>>>(src, dst, rows, K, rowsPad, Kpad);
  };
  auto cvtT = [&](const float* src, unsigned short* dst, int K, int N, int Kpad) {
    long long total = (long long)N * Kpad;
    cvtT_bf16_pad<<<(unsigned)((total + 255) / 256), 256, 0, stream>>>(src, dst, K, N, Kpad);
  };
  auto gemm = [&](const unsigned short* A, const unsigned short* Wt, const float* bias,
                  float* C, int M, int Mpad, int N, int Kpad, int relu) {
    dim3 grid(N / 32, (unsigned)((Mpad / 32 + 3) / 4));
    dim3 block(32, 4);
    wmma_gemm_bf16<<<grid, block, 0, stream>>>(A, Wt, bias, C, M, Mpad, N, Kpad, relu);
  };

  dim3 eblock(32, 4);
  unsigned egrid = (C_E + 3) / 4;

  // -------- input projection: x @ in_w + in_b  (K=177 padded to 192) --------
  cvt(x_in, xbf, C_N, C_IN, C_N, C_INP);
  cvtT(in_w, wbf, C_IN, C_H, C_INP);
  gemm(xbf, wbf, in_b, xcur, C_N, C_N, C_H, C_INP, 0);

  // -------- layers --------
  for (int i = 0; i < C_L; ++i) {
    const float* wei = we + (size_t)i * C_ED * C_H;
    const float* bei = be + (size_t)i * C_H;

    cvt(xcur, xbf, C_N, C_H, C_N, C_H);
    cvtT(wsrc + (size_t)i * C_H * C_H, wbf, C_H, C_H, C_H);
    gemm(xbf, wbf, bsrc + (size_t)i * C_H, hs, C_N, C_N, C_H, C_H, 0);
    cvtT(wdst + (size_t)i * C_H * C_H, wbf, C_H, C_H, C_H);
    gemm(xbf, wbf, bdst + (size_t)i * C_H, hd, C_N, C_N, C_H, C_H, 0);

    hipMemsetAsync(smax, 0, (size_t)C_N * 4, stream);
    hipMemsetAsync(den,  0, (size_t)C_N * 4, stream);
    hipMemsetAsync(xnew, 0, (size_t)C_N * C_H * 4, stream);

    edge_score<<<egrid, eblock, 0, stream>>>(ei, eattr, hs, hd, wei, bei,
                                             avec + (size_t)i * C_H, sE, smax);
    edge_exp<<<(C_E + 255) / 256, 256, 0, stream>>>(ei, sE, smax, den);
    edge_scatter<<<egrid, eblock, 0, stream>>>(ei, eattr, hs, wei, bei, sE, den, xnew);

    rms_kernel<<<C_N, 256, 0, stream>>>(xcur, xnew, normw + (size_t)i * C_H, C_H, 0);

    // FFN
    cvt(xcur, xbf, C_N, C_H, C_N, C_H);
    cvtT(fw1 + (size_t)i * C_H * C_H, wbf, C_H, C_H, C_H);
    gemm(xbf, wbf, fb1 + (size_t)i * C_H, hd, C_N, C_N, C_H, C_H, 1);   // relu
    cvt(hd, xbf, C_N, C_H, C_N, C_H);
    cvtT(fw2 + (size_t)i * C_H * C_H, wbf, C_H, C_H, C_H);
    gemm(xbf, wbf, fb2 + (size_t)i * C_H, hs, C_N, C_N, C_H, C_H, 0);

    rms_kernel<<<C_N, 256, 0, stream>>>(xcur, hs, normw + (size_t)i * C_H, C_H, 0);
  }

  // -------- pooling: z = x @ gap_w + gap_b, att = z @ av_w + av_b, softmax ----
  cvt(xcur, xbf, C_N, C_H, C_N, C_H);
  cvtT(gapw, wbf, C_H, C_H, C_H);
  gemm(xbf, wbf, gapb, hs, C_N, C_N, C_H, C_H, 0);           // hs := z
  att_kernel<<<(C_N + 3) / 4, eblock, 0, stream>>>(hs, avw, avb, att);
  pool_kernel<<<C_B, 256, 0, stream>>>(hs, att, hg);

  // -------- projection head --------
  cvt(hg, xbf, C_B, C_H, C_BPAD, C_H);
  cvtT(pw1, wbf, C_H, C_P, C_H);
  gemm(xbf, wbf, pb1, t1, C_B, C_BPAD, C_P, C_H, 0);
  rms_kernel<<<C_B, 256, 0, stream>>>(t1, nullptr, pnw, C_P, 1);       // rms + relu
  cvt(t1, xbf, C_B, C_P, C_BPAD, C_P);
  cvtT(pw2, wbf, C_P, C_P, C_P);
  gemm(xbf, wbf, pb2, (float*)d_out, C_B, C_BPAD, C_P, C_P, 0);
}
